// NystromLayer_61692910239943
// MI455X (gfx1250) — compile-verified, matrix operand order checked by reading, not run on hardware
//
#include <hip/hip_runtime.h>
#include <hip/hip_bf16.h>
#include <math.h>

// ---------------------------------------------------------------------------
// Problem sizes (fixed by the reference)
// ---------------------------------------------------------------------------
#define B_ROWS 16384
#define S_ROWS 2048
#define D_DIM  512

typedef __bf16 bf16_t;
typedef __attribute__((ext_vector_type(16))) __bf16 bf16x16;
typedef __attribute__((ext_vector_type(8)))  float  f32x8;
typedef __attribute__((ext_vector_type(4)))  int    i32x4;

// Async global->LDS DMA path (CDNA5): guarded so compile never regresses.
#if defined(__gfx1250__) && __has_builtin(__builtin_amdgcn_global_load_async_to_lds_b128) && __has_builtin(__builtin_amdgcn_s_wait_asynccnt)
#define USE_ASYNC_LDS 1
#else
#define USE_ASYNC_LDS 0
#endif

// ---------------------------------------------------------------------------
// bf16 helpers (round-to-nearest via magic add)
// ---------------------------------------------------------------------------
__device__ __forceinline__ bf16_t f2bf(float f) {
    union { float f; unsigned u; } v; v.f = f;
    unsigned r = v.u + 0x7FFFu + ((v.u >> 16) & 1u);
    union { unsigned short s; bf16_t b; } o; o.s = (unsigned short)(r >> 16);
    return o.b;
}
__device__ __forceinline__ float bf2f(bf16_t b) {
    union { unsigned short s; bf16_t b; } i; i.b = b;
    union { unsigned u; float f; } o; o.u = ((unsigned)i.s) << 16;
    return o.f;
}

__device__ __forceinline__ float mishf(float z) {
    // z * tanh(softplus(z)); stable for large z
    float sp = (z > 20.0f) ? z : log1pf(__expf(z));
    return z * tanhf(sp);
}

#if USE_ASYNC_LDS
__device__ __forceinline__ void async_copy16(const bf16_t* g, bf16_t* l) {
    // global_load_async_to_lds_b128: per-lane 16B memory -> LDS, ASYNCcnt.
    // Builtin signature (from clang diagnostic): (v4i AS1*, v4i AS3*, Ii, Ii).
    __builtin_amdgcn_global_load_async_to_lds_b128(
        (__attribute__((address_space(1))) i32x4*)g,
        (__attribute__((address_space(3))) i32x4*)l,
        0, 0);
}
#endif

// ---------------------------------------------------------------------------
// f32 -> bf16 conversion kernel
// ---------------------------------------------------------------------------
__global__ void f32_to_bf16_kernel(const float* __restrict__ in,
                                   bf16_t* __restrict__ out, int n) {
    int i = blockIdx.x * blockDim.x + threadIdx.x;
    int stride = gridDim.x * blockDim.x;
    for (; i < n; i += stride) out[i] = f2bf(in[i]);
}

// ---------------------------------------------------------------------------
// Row sum-of-squares over a bf16 matrix (for RBF distance terms)
// ---------------------------------------------------------------------------
__global__ void rownorm_kernel(const bf16_t* __restrict__ A,
                               float* __restrict__ out, int cols) {
    __shared__ float red[128];
    const int row = blockIdx.x;
    const bf16_t* p = A + (size_t)row * cols;
    float s = 0.f;
    for (int c = threadIdx.x; c < cols; c += blockDim.x) {
        float v = bf2f(p[c]);
        s += v * v;
    }
    red[threadIdx.x] = s;
    __syncthreads();
    for (int off = blockDim.x >> 1; off > 0; off >>= 1) {
        if ((int)threadIdx.x < off) red[threadIdx.x] += red[threadIdx.x + off];
        __syncthreads();
    }
    if (threadIdx.x == 0) out[row] = red[0];
}

// ---------------------------------------------------------------------------
// Tiled WMMA bf16 GEMM:  C[M,N] = epilogue( A[M,K] @ W[N,K]^T )
// Block tile 256(M) x 128(N), K-step 32, 256 threads = 8 waves (4x2).
// Wave tile 64x64 -> 4x4 accumulator frags, 16 WMMA per k-step.
// Double-buffered LDS (48 KB). Tile k+1 is copied memory->LDS with
// GLOBAL_LOAD_ASYNC_TO_LDS_B128 (ASYNCcnt) while tile k's WMMAs run;
// a single s_wait_asynccnt(0) + barrier flips the buffer. No VGPR
// staging -> no scratch spills. Fallback: register-staged copy.
// ---------------------------------------------------------------------------
enum { EPI_MISH_BIAS = 0, EPI_MISH = 1, EPI_BIAS = 2, EPI_RBF = 3, EPI_F32 = 4 };

template <int EPI>
__global__ __launch_bounds__(256, 1)
void gemm_wmma_kernel(const bf16_t* __restrict__ A,     // [M,K] row-major
                      const bf16_t* __restrict__ W,     // [N,K] row-major
                      const float*  __restrict__ bias,  // [N] (bias variants)
                      const float*  __restrict__ rowv,  // [M] (RBF)
                      const float*  __restrict__ colv,  // [N] (RBF)
                      bf16_t* __restrict__ Cb,          // bf16 out
                      float*  __restrict__ Cf,          // f32 out
                      int M, int N, int K) {
    __shared__ __align__(16) bf16_t As[2][256 * 32];  // 16 KB x2
    __shared__ __align__(16) bf16_t Bs[2][128 * 32];  //  8 KB x2

    const int tid  = threadIdx.x;
    const int lane = tid & 31;
    const int wave = tid >> 5;
    const int wm   = wave >> 1;  // 0..3 -> M offset wm*64
    const int wn   = wave & 1;   // 0..1 -> N offset wn*64

    const int bm = blockIdx.y * 256;
    const int bn = blockIdx.x * 128;

    // Copy plan: 16B chunks; A: 256 rows x4 chunks -> 4/thread,
    // B: 128 rows x4 chunks -> 2/thread.
    const int lrow = tid >> 2;       // 0..63
    const int lcol = (tid & 3) * 8;  // bf16 col within 32-wide K slab

    size_t a_base[4];
    size_t b_base[2];
    int lds_a[4], lds_b[2];
#pragma unroll
    for (int q = 0; q < 4; ++q) {
        a_base[q] = (size_t)(bm + lrow + 64 * q) * K + lcol;
        lds_a[q]  = (lrow + 64 * q) * 32 + lcol;
    }
#pragma unroll
    for (int q = 0; q < 2; ++q) {
        b_base[q] = (size_t)(bn + lrow + 64 * q) * K + lcol;
        lds_b[q]  = (lrow + 64 * q) * 32 + lcol;
    }

    f32x8 acc[4][4];
#pragma unroll
    for (int i = 0; i < 4; ++i)
#pragma unroll
        for (int j = 0; j < 4; ++j) { f32x8 z = {}; acc[i][j] = z; }

    const int KT = K >> 5;

#if USE_ASYNC_LDS
    // Prologue: DMA tile 0 into buffer 0.
#pragma unroll
    for (int q = 0; q < 4; ++q) async_copy16(A + a_base[q], &As[0][lds_a[q]]);
#pragma unroll
    for (int q = 0; q < 2; ++q) async_copy16(W + b_base[q], &Bs[0][lds_b[q]]);
    __builtin_amdgcn_s_wait_asynccnt(0);
    __syncthreads();
#else
    uint4 ra[4], rb[2];
#pragma unroll
    for (int q = 0; q < 4; ++q) ra[q] = *(const uint4*)(A + a_base[q]);
#pragma unroll
    for (int q = 0; q < 2; ++q) rb[q] = *(const uint4*)(W + b_base[q]);
#pragma unroll
    for (int q = 0; q < 4; ++q) *(uint4*)(&As[0][lds_a[q]]) = ra[q];
#pragma unroll
    for (int q = 0; q < 2; ++q) *(uint4*)(&Bs[0][lds_b[q]]) = rb[q];
    __syncthreads();
#endif

    // Per-lane fragment addressing per the 16-bit WMMA operand layout:
    // lanes 0-15 hold K 0..7 / 16..23, lanes 16-31 hold K 8..15 / 24..31.
    const int fr   = lane & 15;
    const int half = (lane >> 4) * 8;

    for (int kt = 0; kt < KT; ++kt) {
        const int buf = kt & 1;
        const int nb  = buf ^ 1;

        if (kt + 1 < KT) {  // start copy of next K slab into the other buffer
            const int ko = (kt + 1) << 5;
#if USE_ASYNC_LDS
#pragma unroll
            for (int q = 0; q < 4; ++q)
                async_copy16(A + a_base[q] + ko, &As[nb][lds_a[q]]);
#pragma unroll
            for (int q = 0; q < 2; ++q)
                async_copy16(W + b_base[q] + ko, &Bs[nb][lds_b[q]]);
#else
#pragma unroll
            for (int q = 0; q < 4; ++q) ra[q] = *(const uint4*)(A + a_base[q] + ko);
#pragma unroll
            for (int q = 0; q < 2; ++q) rb[q] = *(const uint4*)(W + b_base[q] + ko);
#endif
        }
        if (kt + 2 < KT) {  // hint the k+2 slab toward L2 (global_prefetch_b8)
            const int ko2 = (kt + 2) << 5;
            __builtin_prefetch(A + a_base[0] + ko2, 0, 1);
            __builtin_prefetch(A + a_base[2] + ko2, 0, 1);
            __builtin_prefetch(W + b_base[0] + ko2, 0, 1);
            __builtin_prefetch(W + b_base[1] + ko2, 0, 1);
        }

        bf16x16 af[4], bfg[4];
#pragma unroll
        for (int j = 0; j < 4; ++j) {
            const bf16_t* p = &Bs[buf][(wn * 64 + j * 16 + fr) * 32 + half];
            union { uint4 q[2]; bf16x16 v; } u;
            u.q[0] = *(const uint4*)(p);
            u.q[1] = *(const uint4*)(p + 16);
            bfg[j] = u.v;
        }
#pragma unroll
        for (int i = 0; i < 4; ++i) {
            const bf16_t* p = &As[buf][(wm * 64 + i * 16 + fr) * 32 + half];
            union { uint4 q[2]; bf16x16 v; } u;
            u.q[0] = *(const uint4*)(p);
            u.q[1] = *(const uint4*)(p + 16);
            af[i] = u.v;
        }

#pragma unroll
        for (int i = 0; i < 4; ++i)
#pragma unroll
            for (int j = 0; j < 4; ++j)
                acc[i][j] = __builtin_amdgcn_wmma_f32_16x16x32_bf16(
                    false, af[i], false, bfg[j], (short)0, acc[i][j],
                    false, false);

        if (kt + 1 < KT) {
#if USE_ASYNC_LDS
            __builtin_amdgcn_s_wait_asynccnt(0);  // own DMA landed in LDS
            __syncthreads();                      // everyone's DMA landed
#else
            __syncthreads();  // everyone done reading the other buffer
#pragma unroll
            for (int q = 0; q < 4; ++q) *(uint4*)(&As[nb][lds_a[q]]) = ra[q];
#pragma unroll
            for (int q = 0; q < 2; ++q) *(uint4*)(&Bs[nb][lds_b[q]]) = rb[q];
            __syncthreads();
#endif
        }
    }

    // Epilogue. C/D frag layout: VGPR v, lane l -> row = v + 8*(l/16), col = l%16.
    const int rhalf = (lane >> 4) * 8;
#pragma unroll
    for (int j = 0; j < 4; ++j) {
        const int col = bn + wn * 64 + j * 16 + fr;
        float bcol = 0.0f, cterm = 0.0f;
        if (EPI == EPI_MISH_BIAS || EPI == EPI_BIAS) bcol = bias[col];
        if (EPI == EPI_RBF) cterm = colv[col];
#pragma unroll
        for (int i = 0; i < 4; ++i) {
#pragma unroll
            for (int v = 0; v < 8; ++v) {
                const int row = bm + wm * 64 + i * 16 + rhalf + v;
                const float a = acc[i][j][v];
                if (EPI == EPI_MISH_BIAS) {
                    Cb[(size_t)row * N + col] = f2bf(mishf(a + bcol));
                } else if (EPI == EPI_MISH) {
                    Cb[(size_t)row * N + col] = f2bf(mishf(a));
                } else if (EPI == EPI_BIAS) {
                    Cb[(size_t)row * N + col] = f2bf(a + bcol);
                } else if (EPI == EPI_RBF) {
                    float d2 = rowv[row] + cterm - 2.0f * a;
                    d2 = d2 > 0.0f ? d2 : 0.0f;
                    Cb[(size_t)row * N + col] = f2bf(__expf(-sqrtf(d2)));
                } else {  // EPI_F32
                    Cf[(size_t)row * N + col] = a;
                }
            }
        }
    }
}

// ---------------------------------------------------------------------------
// Host-side orchestration
// ---------------------------------------------------------------------------
extern "C" void kernel_launch(void* const* d_in, const int* in_sizes, int n_in,
                              void* d_out, int out_size, void* d_ws, size_t ws_size,
                              hipStream_t stream) {
    const float* x   = (const float*)d_in[0];   // [B,D]
    const float* sam = (const float*)d_in[1];   // [S,D]
    const float* W1  = (const float*)d_in[2];   // [D,D]
    const float* b1  = (const float*)d_in[3];   // [D]
    const float* W2  = (const float*)d_in[4];   // [D,D]
    const float* b2  = (const float*)d_in[5];   // [D]
    const float* Wn1 = (const float*)d_in[6];   // [S,S]
    const float* Wn2 = (const float*)d_in[7];   // [S,S]
    float* out = (float*)d_out;                 // [B,S]

    char* ws = (char*)d_ws;
    size_t off = 0;
    auto alloc = [&](size_t bytes) -> char* {
        char* p = ws + off;
        off += (bytes + 255) & ~(size_t)255;
        return p;
    };

    bf16_t* x_bf   = (bf16_t*)alloc((size_t)B_ROWS * D_DIM * 2);
    bf16_t* s_bf   = (bf16_t*)alloc((size_t)S_ROWS * D_DIM * 2);
    bf16_t* W1_bf  = (bf16_t*)alloc((size_t)D_DIM * D_DIM * 2);
    bf16_t* W2_bf  = (bf16_t*)alloc((size_t)D_DIM * D_DIM * 2);
    bf16_t* Wn1_bf = (bf16_t*)alloc((size_t)S_ROWS * S_ROWS * 2);
    bf16_t* Wn2_bf = (bf16_t*)alloc((size_t)S_ROWS * S_ROWS * 2);
    bf16_t* H1x    = (bf16_t*)alloc((size_t)B_ROWS * D_DIM * 2);
    bf16_t* Xd     = (bf16_t*)alloc((size_t)B_ROWS * D_DIM * 2);
    bf16_t* H1s    = (bf16_t*)alloc((size_t)S_ROWS * D_DIM * 2);
    bf16_t* Sd     = (bf16_t*)alloc((size_t)S_ROWS * D_DIM * 2);
    float*  xx     = (float*)alloc((size_t)B_ROWS * 4);
    float*  ssq    = (float*)alloc((size_t)S_ROWS * 4);
    bf16_t* Km     = (bf16_t*)alloc((size_t)B_ROWS * S_ROWS * 2);
    bf16_t* Hm     = (bf16_t*)alloc((size_t)B_ROWS * S_ROWS * 2);

    auto conv = [&](const float* in, bf16_t* o, size_t n) {
        int blocks = (int)((n + 255) / 256);
        if (blocks > 8192) blocks = 8192;
        f32_to_bf16_kernel<<<blocks, 256, 0, stream>>>(in, o, (int)n);
    };
    conv(x,   x_bf,   (size_t)B_ROWS * D_DIM);
    conv(sam, s_bf,   (size_t)S_ROWS * D_DIM);
    conv(W1,  W1_bf,  (size_t)D_DIM * D_DIM);
    conv(W2,  W2_bf,  (size_t)D_DIM * D_DIM);
    conv(Wn1, Wn1_bf, (size_t)S_ROWS * S_ROWS);
    conv(Wn2, Wn2_bf, (size_t)S_ROWS * S_ROWS);

    const dim3 blk(256);
    const dim3 gD_B(D_DIM / 128, B_ROWS / 256);   // [B,D] outputs
    const dim3 gD_S(D_DIM / 128, S_ROWS / 256);   // [S,D] outputs
    const dim3 gS_B(S_ROWS / 128, B_ROWS / 256);  // [B,S] outputs

    // dml(x): H1x = mish(x @ W1^T + b1); Xd = H1x @ W2^T + b2
    gemm_wmma_kernel<EPI_MISH_BIAS><<<gD_B, blk, 0, stream>>>(
        x_bf, W1_bf, b1, nullptr, nullptr, H1x, nullptr, B_ROWS, D_DIM, D_DIM);
    gemm_wmma_kernel<EPI_BIAS><<<gD_B, blk, 0, stream>>>(
        H1x, W2_bf, b2, nullptr, nullptr, Xd, nullptr, B_ROWS, D_DIM, D_DIM);

    // dml(samples)
    gemm_wmma_kernel<EPI_MISH_BIAS><<<gD_S, blk, 0, stream>>>(
        s_bf, W1_bf, b1, nullptr, nullptr, H1s, nullptr, S_ROWS, D_DIM, D_DIM);
    gemm_wmma_kernel<EPI_BIAS><<<gD_S, blk, 0, stream>>>(
        H1s, W2_bf, b2, nullptr, nullptr, Sd, nullptr, S_ROWS, D_DIM, D_DIM);

    // Row norms for the RBF distance
    rownorm_kernel<<<B_ROWS, 128, 0, stream>>>(Xd, xx,  D_DIM);
    rownorm_kernel<<<S_ROWS, 128, 0, stream>>>(Sd, ssq, D_DIM);

    // K = exp(-sqrt(max(||x||^2 + ||s||^2 - 2 x.s, 0)))  fused into the GEMM
    gemm_wmma_kernel<EPI_RBF><<<gS_B, blk, 0, stream>>>(
        Xd, Sd, nullptr, xx, ssq, Km, nullptr, B_ROWS, S_ROWS, D_DIM);

    // H = mish(K @ Wn1^T)   (no bias)
    gemm_wmma_kernel<EPI_MISH><<<gS_B, blk, 0, stream>>>(
        Km, Wn1_bf, nullptr, nullptr, nullptr, Hm, nullptr, B_ROWS, S_ROWS, S_ROWS);

    // out = H @ Wn2^T  (f32 output)
    gemm_wmma_kernel<EPI_F32><<<gS_B, blk, 0, stream>>>(
        Hm, Wn2_bf, nullptr, nullptr, nullptr, nullptr, out, B_ROWS, S_ROWS, S_ROWS);

    (void)in_sizes; (void)n_in; (void)out_size; (void)ws_size;
}
